// XTAttention_56092272886434
// MI455X (gfx1250) — matrix operand order repaired and checked by reading, not compiled, not run on hardware
//
#include <hip/hip_runtime.h>

// ---------------------------------------------------------------------------
// MI455X (gfx1250, wave32) WMMA implementation of 3-stream cross attention.
// All GEMM work via v_wmma_f32_16x16x32_f16 (f16 in, f32 acc).
// Pipeline: proj_gemm (7 fused projections, 128x128 tiles)
//           vt_kernel (one-shot V transpose to [bh][d][m])
//           attn_kernel (flash attention, BQ=128, BK=64, Q in registers)
//           out_gemm (final projection + bias, fp32 out)
// ---------------------------------------------------------------------------

typedef __attribute__((ext_vector_type(16))) _Float16 v16h;
typedef __attribute__((ext_vector_type(8)))  _Float16 v8h;
typedef __attribute__((ext_vector_type(2)))  __fp16   v2hf;  // cvt_pkrtz result type
typedef __attribute__((ext_vector_type(8)))  float    v8f;

#define HEADS 16
#define HDIM  64
#define DIMX  1024
#define BATCH 4
#define NSEQ  2048
#define MSEQ  2048
#define DK3   192   // concatenated head dim (3 * 64)
#define BQ    128   // query rows per attention block
#define BKK   64    // keys per attention chunk

__device__ __forceinline__ v16h combine16(v8h lo, v8h hi) {
  return __builtin_shufflevector(lo, hi, 0, 1, 2, 3, 4, 5, 6, 7,
                                 8, 9, 10, 11, 12, 13, 14, 15);
}

// Pack two fp32 to f16 pair (v_cvt_pk_rtz_f16_f32) and store as 4 bytes.
__device__ __forceinline__ void store_pk(_Float16* dst, float a, float b) {
  const v2hf h = __builtin_amdgcn_cvt_pkrtz(a, b);
  *(v2hf*)dst = h;
}

// A operand, 16x32 f16 (MxK), LDS tile [row][k].
// Lane l: row = l&15; k base (l>>4)*8, halves [0..7] then [16..23].
// Requires 16B alignment: ld*2, k0*2 multiples of 16 bytes.
__device__ __forceinline__ v16h load_frag_a(const _Float16* base, int ld,
                                            int row0, int k0) {
  const int lane = threadIdx.x & 31;
  const _Float16* p = base + (row0 + (lane & 15)) * ld + k0 + ((lane >> 4) << 3);
  const v8h lo = *(const v8h*)p;
  const v8h hi = *(const v8h*)(p + 16);
  return combine16(lo, hi);
}

// B operand, 32x16 f16 (KxN), LDS tile stored [col][k].
// Lane l: col = l&15; k base (l>>4)*8, halves [0..7] then [16..23].
__device__ __forceinline__ v16h load_frag_b(const _Float16* base, int ld,
                                            int col0, int k0) {
  const int lane = threadIdx.x & 31;
  const _Float16* p = base + (col0 + (lane & 15)) * ld + k0 + ((lane >> 4) << 3);
  const v8h lo = *(const v8h*)p;
  const v8h hi = *(const v8h*)(p + 16);
  return combine16(lo, hi);
}

__device__ __forceinline__ v8f wmma16(v16h a, v16h b, v8f c) {
  return __builtin_amdgcn_wmma_f32_16x16x32_f16(
      false, a, false, b, (short)0, c, false, false);
}

// ---------------------------------------------------------------------------
// Kernel 1: all seven input projections (fp32 in, f16 head-split out).
// 128x128 tile, 256 threads = 8 waves (2x4), each wave 64x32 via 4x2 WMMA.
// ---------------------------------------------------------------------------
struct ProjArgs {
  const float* x[7];
  const float* w[7];
};

__global__ __launch_bounds__(256)
void proj_gemm_kernel(ProjArgs args, _Float16* __restrict__ Qcat,
                      _Float16* __restrict__ Kcat, _Float16* __restrict__ Vbuf) {
  __shared__ __align__(16) _Float16 As[128 * 40];  // [row][k], ld 40
  __shared__ __align__(16) _Float16 Bs[128 * 40];  // [col][k], ld 40

  const int z    = blockIdx.z;
  const float* X = args.x[z];
  const float* W = args.w[z];
  const int tid  = threadIdx.x;
  const int lane = tid & 31;
  const int wave = tid >> 5;
  const int m0w  = (wave >> 2) * 64;
  const int n0w  = (wave & 3) * 32;
  const int row0 = blockIdx.y * 128;
  const int col0 = blockIdx.x * 128;

  v8f acc[4][2] = {};

  for (int kb = 0; kb < DIMX; kb += 32) {
#pragma unroll
    for (int e = 0; e < 8; ++e) {
      const int ia = e * 256 + tid;            // 0..2047 pair index
      // A: row ra, k pair ka (coalesced float2 over k)
      const int ra = ia >> 4;
      const int ka = (ia & 15) * 2;
      const float2 xa = *(const float2*)&X[(size_t)(row0 + ra) * DIMX + kb + ka];
      store_pk(&As[ra * 40 + ka], xa.x, xa.y);
      // B: col cb (coalesced over n), k pair kp (two strided loads, packed)
      const int cb = ia & 127;
      const int kp = (ia >> 7) * 2;
      const float w0 = W[(size_t)(kb + kp) * DIMX + col0 + cb];
      const float w1 = W[(size_t)(kb + kp + 1) * DIMX + col0 + cb];
      store_pk(&Bs[cb * 40 + kp], w0, w1);
    }
    __syncthreads();
    v16h af[4];
    v16h bf[2];
#pragma unroll
    for (int ti = 0; ti < 4; ++ti) af[ti] = load_frag_a(As, 40, m0w + ti * 16, 0);
#pragma unroll
    for (int tj = 0; tj < 2; ++tj) bf[tj] = load_frag_b(Bs, 40, n0w + tj * 16, 0);
#pragma unroll
    for (int ti = 0; ti < 4; ++ti)
#pragma unroll
      for (int tj = 0; tj < 2; ++tj)
        acc[ti][tj] = wmma16(af[ti], bf[tj], acc[ti][tj]);
    __syncthreads();
  }

  // epilogue: scatter into head-split layouts
  const int stream = (z < 3) ? z : (z < 6 ? (z - 3) : 0);
  const int cloc = lane & 15;
  const int mloc = (lane >> 4) << 3;
#pragma unroll
  for (int ti = 0; ti < 4; ++ti) {
#pragma unroll
    for (int tj = 0; tj < 2; ++tj) {
      const int c  = col0 + n0w + tj * 16 + cloc;
      const int hh = c >> 6;
      const int dd = c & 63;
#pragma unroll
      for (int i = 0; i < 8; ++i) {
        const int r  = row0 + m0w + ti * 16 + mloc + i;
        const int bb = r >> 11;
        const int nn = r & 2047;
        const _Float16 val = (_Float16)acc[ti][tj][i];
        if (z < 6) {
          _Float16* D = (z < 3) ? Qcat : Kcat;
          D[((size_t)(bb * HEADS + hh) * NSEQ + nn) * DK3 + stream * HDIM + dd] = val;
        } else {
          Vbuf[((size_t)(bb * HEADS + hh) * MSEQ + nn) * HDIM + dd] = val;
        }
      }
    }
  }
}

// ---------------------------------------------------------------------------
// Kernel 1b: one-shot V transpose  [bh][m][d] -> [bh][d][m]
// ---------------------------------------------------------------------------
__global__ __launch_bounds__(256)
void vt_kernel(const _Float16* __restrict__ Vbuf, _Float16* __restrict__ Vt) {
  __shared__ __align__(16) _Float16 T[64 * 72];  // [d][m], ld 72
  const int tid = threadIdx.x;
  const int bh  = blockIdx.y;
  const int m0  = blockIdx.x * 64;
  const _Float16* src = Vbuf + ((size_t)bh * MSEQ + m0) * HDIM;

  // load 64x64 tile (uint4 = 8 halves over d), scatter-transpose into LDS
#pragma unroll
  for (int e = 0; e < 2; ++e) {
    const int idx = e * 256 + tid;       // 0..511
    const int m = idx >> 3;
    const int d = (idx & 7) * 8;
    uint4 v4 = *(const uint4*)&src[(size_t)m * HDIM + d];
    const _Float16* hv = (const _Float16*)&v4;
#pragma unroll
    for (int q = 0; q < 8; ++q) T[(d + q) * 72 + m] = hv[q];
  }
  __syncthreads();
  // write rows of Vt (coalesced, vectorized)
#pragma unroll
  for (int e = 0; e < 2; ++e) {
    const int idx = e * 256 + tid;
    const int d = idx >> 3;
    const int q = (idx & 7) * 8;
    *(uint4*)&Vt[((size_t)bh * HDIM + d) * MSEQ + m0 + q] = *(const uint4*)&T[d * 72 + q];
  }
}

// ---------------------------------------------------------------------------
// Kernel 2: flash attention per (b, h). 256 threads = 8 waves, 16 q-rows each.
// Q fragments pre-scaled and held in registers; K staged 64 keys at a time;
// V read from transposed layout with single vector copy.
// ---------------------------------------------------------------------------
__global__ __launch_bounds__(256)
void attn_kernel(const _Float16* __restrict__ Qcat,
                 const _Float16* __restrict__ Kcat,
                 const _Float16* __restrict__ Vt,
                 _Float16* __restrict__ AttnOut) {
  __shared__ __align__(16) _Float16 Ks[64 * 200];      // [key][k], ld 200
  __shared__ __align__(16) _Float16 Vs[64 * 72];       // [d][m],  ld 72
  __shared__ __align__(16) _Float16 Ps[8 * 16 * 72];   // per-wave P [row][m]

  const int tid  = threadIdx.x;
  const int lane = tid & 31;
  const int wave = tid >> 5;
  const int b  = blockIdx.z;
  const int h  = blockIdx.y;
  const int qt = blockIdx.x;

  const size_t bh = (size_t)(b * HEADS + h);
  const _Float16* Kp = Kcat + bh * MSEQ * DK3;
  const _Float16* Vp = Vt + bh * (size_t)HDIM * MSEQ;
  const int waveRow = wave * 16;

  // Q fragments in registers, pre-scaled by d^-0.5 = 0.125 (exact in f16)
  v16h qf[6];
  {
    const _Float16* qrow =
        Qcat + (bh * NSEQ + (size_t)qt * BQ + waveRow + (lane & 15)) * DK3 +
        ((lane >> 4) << 3);
    const _Float16 s = (_Float16)0.125f;
#pragma unroll
    for (int kk = 0; kk < 6; ++kk) {
      v8h lo = *(const v8h*)(qrow + kk * 32);
      v8h hi = *(const v8h*)(qrow + kk * 32 + 16);
      lo = lo * s;
      hi = hi * s;
      qf[kk] = combine16(lo, hi);
    }
  }

  v8f oacc[4] = {};
  float mrow[8], lrow[8];
#pragma unroll
  for (int i = 0; i < 8; ++i) { mrow[i] = -1e30f; lrow[i] = 0.0f; }

  _Float16* Pw = Ps + wave * 16 * 72;
  const int mloc = (lane >> 4) << 3;
  const int cloc = lane & 15;

  for (int j0 = 0; j0 < MSEQ; j0 += BKK) {
    __syncthreads();  // protect Ks/Vs/Ps reuse across iterations
    // stage K chunk: 64 rows x 192 halves, 16B vector copies
    for (int idx = tid; idx < 64 * 24; idx += 256) {
      const int r = idx / 24;
      const int q = (idx - r * 24) * 8;
      *(uint4*)&Ks[r * 200 + q] = *(const uint4*)&Kp[(size_t)(j0 + r) * DK3 + q];
    }
    // stage V chunk from transposed layout: 64 d-rows x 64 halves
#pragma unroll
    for (int e = 0; e < 2; ++e) {
      const int idx = e * 256 + tid;
      const int d = idx >> 3;
      const int q = (idx & 7) * 8;
      *(uint4*)&Vs[d * 72 + q] = *(const uint4*)&Vp[(size_t)d * MSEQ + j0 + q];
    }
    // prefetch next K chunk into cache (global_prefetch_b8)
    if (j0 + BKK < MSEQ) {
      __builtin_prefetch(Kp + (size_t)(j0 + BKK) * DK3 + (size_t)tid * 48, 0, 0);
    }
    __syncthreads();

    // S = Q K^T : 4 tiles of 16x16, contraction over d=192
    v8f s0 = {}, s1 = {}, s2 = {}, s3 = {};
#pragma unroll
    for (int kk = 0; kk < 6; ++kk) {
      const v16h b0 = load_frag_b(Ks, 200, 0, kk * 32);
      const v16h b1 = load_frag_b(Ks, 200, 16, kk * 32);
      const v16h b2 = load_frag_b(Ks, 200, 32, kk * 32);
      const v16h b3 = load_frag_b(Ks, 200, 48, kk * 32);
      s0 = wmma16(qf[kk], b0, s0);
      s1 = wmma16(qf[kk], b1, s1);
      s2 = wmma16(qf[kk], b2, s2);
      s3 = wmma16(qf[kk], b3, s3);
    }

    // online softmax (row = mloc+i, spread over 16 lanes of a wave half)
#pragma unroll
    for (int i = 0; i < 8; ++i) {
      const float v0 = s0[i], v1 = s1[i], v2 = s2[i], v3 = s3[i];
      float rmax = fmaxf(fmaxf(v0, v1), fmaxf(v2, v3));
#pragma unroll
      for (int off = 1; off < 16; off <<= 1)
        rmax = fmaxf(rmax, __shfl_xor(rmax, off, 32));
      const float mnew  = fmaxf(mrow[i], rmax);
      const float alpha = __expf(mrow[i] - mnew);
      const float p0 = __expf(v0 - mnew);
      const float p1 = __expf(v1 - mnew);
      const float p2 = __expf(v2 - mnew);
      const float p3 = __expf(v3 - mnew);
      float rs = (p0 + p1) + (p2 + p3);
#pragma unroll
      for (int off = 1; off < 16; off <<= 1) rs += __shfl_xor(rs, off, 32);
      lrow[i] = lrow[i] * alpha + rs;
      mrow[i] = mnew;
#pragma unroll
      for (int dt = 0; dt < 4; ++dt) oacc[dt][i] *= alpha;
      Pw[(mloc + i) * 72 + cloc]      = (_Float16)p0;
      Pw[(mloc + i) * 72 + 16 + cloc] = (_Float16)p1;
      Pw[(mloc + i) * 72 + 32 + cloc] = (_Float16)p2;
      Pw[(mloc + i) * 72 + 48 + cloc] = (_Float16)p3;
    }
    __syncthreads();  // make P visible across lanes in A-operand layout

    // O += P V : contraction over 64 keys (two 32-k steps), 4 d-tiles
    const v16h ap0 = load_frag_a(Pw, 72, 0, 0);
    const v16h ap1 = load_frag_a(Pw, 72, 0, 32);
#pragma unroll
    for (int dt = 0; dt < 4; ++dt) {
      const v16h bv0 = load_frag_b(Vs, 72, dt * 16, 0);
      const v16h bv1 = load_frag_b(Vs, 72, dt * 16, 32);
      oacc[dt] = wmma16(ap0, bv0, oacc[dt]);
      oacc[dt] = wmma16(ap1, bv1, oacc[dt]);
    }
  }

  // epilogue: normalize + write f16 [b*N + n][h*64 + d]
  const int rowG = b * NSEQ + qt * BQ + waveRow + mloc;
  const int colB = h * HDIM + cloc;
#pragma unroll
  for (int i = 0; i < 8; ++i) {
    const float inv = 1.0f / lrow[i];
    const size_t rb = (size_t)(rowG + i) * DIMX;
#pragma unroll
    for (int dt = 0; dt < 4; ++dt)
      AttnOut[rb + colB + dt * 16] = (_Float16)(oacc[dt][i] * inv);
  }
}

// ---------------------------------------------------------------------------
// Kernel 3: output projection (f16 attn @ Wo + bo -> fp32 d_out), 128x128 tile
// ---------------------------------------------------------------------------
__global__ __launch_bounds__(256)
void out_gemm_kernel(const _Float16* __restrict__ A,
                     const float* __restrict__ Wo,
                     const float* __restrict__ bo,
                     float* __restrict__ Out) {
  __shared__ __align__(16) _Float16 As[128 * 40];
  __shared__ __align__(16) _Float16 Bs[128 * 40];

  const int tid  = threadIdx.x;
  const int lane = tid & 31;
  const int wave = tid >> 5;
  const int m0w  = (wave >> 2) * 64;
  const int n0w  = (wave & 3) * 32;
  const int row0 = blockIdx.y * 128;
  const int col0 = blockIdx.x * 128;

  v8f acc[4][2] = {};

  for (int kb = 0; kb < DIMX; kb += 32) {
    // A: f16 source, straight 16B vector copies (128 rows x 4 chunks)
#pragma unroll
    for (int e = 0; e < 2; ++e) {
      const int idx = e * 256 + tid;
      const int r = idx >> 2;
      const int q = (idx & 3) * 8;
      *(uint4*)&As[r * 40 + q] = *(const uint4*)&A[(size_t)(row0 + r) * DIMX + kb + q];
    }
    // B: fp32 source, packed convert
#pragma unroll
    for (int e = 0; e < 8; ++e) {
      const int ib = e * 256 + tid;
      const int cb = ib & 127;
      const int kp = (ib >> 7) * 2;
      const float w0 = Wo[(size_t)(kb + kp) * DIMX + col0 + cb];
      const float w1 = Wo[(size_t)(kb + kp + 1) * DIMX + col0 + cb];
      store_pk(&Bs[cb * 40 + kp], w0, w1);
    }
    __syncthreads();
    v16h af[4];
    v16h bf[2];
#pragma unroll
    for (int ti = 0; ti < 4; ++ti) af[ti] = load_frag_a(As, 40, m0w + ti * 16, 0);
#pragma unroll
    for (int tj = 0; tj < 2; ++tj) bf[tj] = load_frag_b(Bs, 40, n0w + tj * 16, 0);
#pragma unroll
    for (int ti = 0; ti < 4; ++ti)
#pragma unroll
      for (int tj = 0; tj < 2; ++tj)
        acc[ti][tj] = wmma16(af[ti], bf[tj], acc[ti][tj]);
    __syncthreads();
  }

  const int cloc = lane & 15;
  const int mloc = (lane >> 4) << 3;
#pragma unroll
  for (int tj = 0; tj < 2; ++tj) {
    const int c = col0 + n0w + tj * 16 + cloc;
    const float bias = bo[c];
#pragma unroll
    for (int ti = 0; ti < 4; ++ti) {
#pragma unroll
      for (int i = 0; i < 8; ++i) {
        const int r = row0 + m0w + ti * 16 + mloc + i;
        Out[(size_t)r * DIMX + c] = acc[ti][tj][i] + bias;
      }
    }
  }
}

// ---------------------------------------------------------------------------
extern "C" void kernel_launch(void* const* d_in, const int* in_sizes, int n_in,
                              void* d_out, int out_size, void* d_ws,
                              size_t ws_size, hipStream_t stream) {
  (void)in_sizes; (void)n_in; (void)out_size; (void)ws_size;

  const float* words           = (const float*)d_in[0];
  const float* position_words  = (const float*)d_in[1];
  const float* conscious_words = (const float*)d_in[2];
  const float* cross_kv        = (const float*)d_in[3];
  const float* position_cross  = (const float*)d_in[4];
  const float* conscious_cross = (const float*)d_in[5];
  const float* Wwq = (const float*)d_in[6];
  const float* Wwk = (const float*)d_in[7];
  const float* Wwv = (const float*)d_in[8];
  const float* Wpq = (const float*)d_in[9];
  const float* Wpk = (const float*)d_in[10];
  const float* Wcq = (const float*)d_in[11];
  const float* Wck = (const float*)d_in[12];
  const float* Wo  = (const float*)d_in[13];
  const float* bo  = (const float*)d_in[14];

  // Workspace (f16): Qcat 48MB | Kcat 48MB | V 16MB | Vt 16MB | AttnOut 16MB
  _Float16* Qcat = (_Float16*)d_ws;
  _Float16* Kcat = Qcat + (size_t)BATCH * HEADS * NSEQ * DK3;
  _Float16* Vbuf = Kcat + (size_t)BATCH * HEADS * MSEQ * DK3;
  _Float16* Vt   = Vbuf + (size_t)BATCH * HEADS * MSEQ * HDIM;
  _Float16* Attn = Vt   + (size_t)BATCH * HEADS * MSEQ * HDIM;

  ProjArgs pa;
  pa.x[0] = words;           pa.w[0] = Wwq;
  pa.x[1] = position_words;  pa.w[1] = Wpq;
  pa.x[2] = conscious_words; pa.w[2] = Wcq;
  pa.x[3] = cross_kv;        pa.w[3] = Wwk;
  pa.x[4] = position_cross;  pa.w[4] = Wpk;
  pa.x[5] = conscious_cross; pa.w[5] = Wck;
  pa.x[6] = cross_kv;        pa.w[6] = Wwv;

  dim3 g1(DIMX / 128, (BATCH * NSEQ) / 128, 7);
  proj_gemm_kernel<<<g1, 256, 0, stream>>>(pa, Qcat, Kcat, Vbuf);

  dim3 gv(MSEQ / 64, BATCH * HEADS);
  vt_kernel<<<gv, 256, 0, stream>>>(Vbuf, Vt);

  dim3 g2(NSEQ / BQ, HEADS, BATCH);
  attn_kernel<<<g2, 256, 0, stream>>>(Qcat, Kcat, Vt, Attn);

  dim3 g3(DIMX / 128, (BATCH * NSEQ) / 128);
  out_gemm_kernel<<<g3, 256, 0, stream>>>(Attn, Wo, bo, (float*)d_out);
}
